// SympleAgent_27273042330323
// MI455X (gfx1250) — matrix operand order repaired and checked by reading, not compiled, not run on hardware
//
#include <hip/hip_runtime.h>
#include <hip/hip_bf16.h>
#include <math.h>

typedef __bf16 bf16_t;
typedef __attribute__((ext_vector_type(16))) __bf16 v16bf;
typedef __attribute__((ext_vector_type(8)))  float  v8f;

#define NTREE 4095      // 2^12 - 1
#define BATCH 128
#define HDIM  64
#define ZDIM  320       // 5*H

__device__ __forceinline__ float sigf(float x) { return 1.0f / (1.0f + __expf(-x)); }

// ---------------------------------------------------------------------------
// Pack Ua/Ub (f32, [K=64][N=320] row-major) into the CDNA5 16-bit B-matrix
// fragment layout: pw[((mat*20+nt)*2+kh)*32 + lane][16 bf16].
// For lane l: col = nt*16 + (l&15), g = l>>4; element t:
// K = kh*32 + 16*(t/8) + 8*g + (t%8).  (2 matrices * 20 n-tiles * 2 k-halves)
// ---------------------------------------------------------------------------
__global__ void pack_weights(const float* __restrict__ Ua,
                             const float* __restrict__ Ub,
                             bf16_t* __restrict__ pw) {
    int tid = blockIdx.x * 256 + threadIdx.x;          // 0..40959
    int t    = tid & 15;
    int lane = (tid >> 4) & 31;
    int kh   = (tid >> 9) & 1;
    int nt   = (tid >> 10) % 20;
    int mat  = (tid >> 10) / 20;
    const float* W = mat ? Ub : Ua;
    int col = nt * 16 + (lane & 15);
    int g   = lane >> 4;
    int v   = t >> 1, e = t & 1;
    int K   = kh * 32 + (t >> 3) * 16 + g * 8 + (t & 7);
    pw[tid] = (bf16_t)W[K * ZDIM + col];
}

// ---------------------------------------------------------------------------
// Leaf level (d=11): z = Wx[idx] + b_tree; no children, so only i/o/u gates.
// One thread per (row, h): rows = b*2048 + j, 262144 rows total.
// ---------------------------------------------------------------------------
__global__ void tree_leaf(const int* __restrict__ x_idx,
                          const float* __restrict__ Wx,
                          const float* __restrict__ b_tree,
                          bf16_t* __restrict__ h_out,
                          float*  __restrict__ c_out) {
    size_t gid = (size_t)blockIdx.x * 256 + threadIdx.x;   // 262144*64 threads
    int    hh  = (int)(gid & 63);
    size_t row = gid >> 6;                                  // b*2048 + j
    int    b   = (int)(row >> 11);
    int    j   = (int)(row & 2047);
    int    idx = x_idx[b * NTREE + 2047 + j];
    const float* wrow = Wx + (size_t)idx * ZDIM;
    float zi = wrow[hh]        + b_tree[hh];
    float zo = wrow[192 + hh]  + b_tree[192 + hh];
    float zu = wrow[256 + hh]  + b_tree[256 + hh];
    float c  = sigf(zi) * tanhf(zu);
    float h  = sigf(zo) * tanhf(c);
    c_out[gid] = c;
    h_out[gid] = (bf16_t)h;
}

// ---------------------------------------------------------------------------
// Internal level: one workgroup (128 threads / 4 waves) per 32-row block
// (2 M-tiles of 16). Wave w owns N-tiles {w, w+4, w+8, w+12, w+16}, i.e.
// columns 16w..16w+15 of ALL FIVE gate chunks -> the five gate accumulators
// stay in registers and the cell update fuses directly onto the WMMA output
// (no z round-trip through LDS). B fragments are reused across both M-tiles.
// Children of level-row rg are rows 2rg, 2rg+1 of the previous level.
// ---------------------------------------------------------------------------
__global__ void __launch_bounds__(128)
tree_level_wmma(int dl,                       // log2(n)
                const int*   __restrict__ x_idx,
                const float* __restrict__ Wx,
                const float* __restrict__ b_tree,
                const bf16_t* __restrict__ packW,
                const bf16_t* __restrict__ h_prev,
                const float*  __restrict__ c_prev,
                bf16_t* __restrict__ h_out,
                float*  __restrict__ c_out) {
    // a_frag[mtile(2)][mat(2)][kh(2)][lane(32)][16 bf16] = 8 KB
    __shared__ __align__(32) bf16_t a_frag[2 * 2 * 2 * 32 * 16];

    const int tid = threadIdx.x;           // 0..127
    const int m0  = blockIdx.x * 32;       // first output row of this block
    const int n   = 1 << dl;

    // ---- stage A: 2 mtiles x (16 left + 16 right child rows) -> swizzled ----
    // 16-bit A 16x32 layout: lane = g*16 + row; VGPR v<4: K=g*8+2v+e,
    // v>=4: K=16+g*8+2(v-4)+e  (per 32-wide K half kh).
    for (int q = 0; q < 32; ++q) {
        int li   = tid * 32 + q;           // 0..4095
        int mt   = li >> 11;
        int rem  = li & 2047;
        int mat  = rem >> 10;              // 0: h_a (even child), 1: h_b (odd)
        int rem2 = rem & 1023;
        int r    = rem2 >> 6, k = rem2 & 63;
        bf16_t val = h_prev[(size_t)(2 * (m0 + mt * 16 + r) + mat) * HDIM + k];
        int kh = k >> 5, kr = k & 31;
        int g, v, e = kr & 1;
        if (kr < 16) { g = kr >> 3;              v = (kr & 7) >> 1; }
        else         { int kk = kr - 16; g = kk >> 3; v = 4 + ((kk & 7) >> 1); }
        int lane = g * 16 + r;
        a_frag[((mt * 4 + mat * 2 + kh) * 32 + lane) * 16 + (v * 2 + e)] = val;
    }
    __syncthreads();

    // ---- WMMA: per gate chunk g, wave owns N-tile nt = g*4 + wave ----
    const int wave = tid >> 5;
    const int lane = tid & 31;
    const v16bf* af = (const v16bf*)a_frag;
    const v16bf* pb = (const v16bf*)packW;

    v8f acc[2][5];
    v8f zero = {};
#pragma unroll
    for (int mt = 0; mt < 2; ++mt)
#pragma unroll
        for (int g = 0; g < 5; ++g) acc[mt][g] = zero;

#pragma unroll
    for (int g = 0; g < 5; ++g) {
        int nt = g * 4 + wave;
        v16bf bua0 = pb[((0 * 20 + nt) * 2 + 0) * 32 + lane];
        v16bf bua1 = pb[((0 * 20 + nt) * 2 + 1) * 32 + lane];
        v16bf bub0 = pb[((1 * 20 + nt) * 2 + 0) * 32 + lane];
        v16bf bub1 = pb[((1 * 20 + nt) * 2 + 1) * 32 + lane];
#pragma unroll
        for (int mt = 0; mt < 2; ++mt) {
            v16bf aa0 = af[(mt * 4 + 0) * 32 + lane];
            v16bf aa1 = af[(mt * 4 + 1) * 32 + lane];
            v16bf ab0 = af[(mt * 4 + 2) * 32 + lane];
            v16bf ab1 = af[(mt * 4 + 3) * 32 + lane];
            v8f a = acc[mt][g];
            a = __builtin_amdgcn_wmma_f32_16x16x32_bf16(false, aa0, false, bua0,
                                                        (short)0, a, false, false);
            a = __builtin_amdgcn_wmma_f32_16x16x32_bf16(false, aa1, false, bua1,
                                                        (short)0, a, false, false);
            a = __builtin_amdgcn_wmma_f32_16x16x32_bf16(false, ab0, false, bub0,
                                                        (short)0, a, false, false);
            a = __builtin_amdgcn_wmma_f32_16x16x32_bf16(false, ab1, false, bub1,
                                                        (short)0, a, false, false);
            acc[mt][g] = a;
        }
    }

    // ---- fused gate phase straight out of the accumulators ----
    // C/D layout: VGPR v, lane l -> row = v + 8*(l>>4), col = l&15.
    // acc[mt][0..4] = i, fa, fb, o, u for column hh = wave*16 + (l&15).
    const int hh = wave * 16 + (lane & 15);
#pragma unroll
    for (int mt = 0; mt < 2; ++mt) {
#pragma unroll
        for (int v = 0; v < 8; ++v) {
            int r  = mt * 16 + v + ((lane >> 4) << 3);
            int rg = m0 + r;
            int b  = rg >> dl;
            int j  = rg & (n - 1);
            int idx = x_idx[b * NTREE + (n - 1) + j];
            const float* wrow = Wx + (size_t)idx * ZDIM;
            float zi  = acc[mt][0][v] + wrow[hh]       + b_tree[hh];
            float zfa = acc[mt][1][v] + wrow[64 + hh]  + b_tree[64 + hh];
            float zfb = acc[mt][2][v] + wrow[128 + hh] + b_tree[128 + hh];
            float zo  = acc[mt][3][v] + wrow[192 + hh] + b_tree[192 + hh];
            float zu  = acc[mt][4][v] + wrow[256 + hh] + b_tree[256 + hh];
            float ca  = c_prev[(size_t)(2 * rg)     * HDIM + hh];
            float cb  = c_prev[(size_t)(2 * rg + 1) * HDIM + hh];
            float c = sigf(zi) * tanhf(zu) + sigf(zfa) * ca + sigf(zfb) * cb;
            float h = sigf(zo) * tanhf(c);
            c_out[(size_t)rg * HDIM + hh] = c;
            h_out[(size_t)rg * HDIM + hh] = (bf16_t)h;
        }
    }
}

// ---------------------------------------------------------------------------
// Head: f = relu(h_root@W1+b1)@W2+b2; LSTM cell with h0=0 (W_hh unused);
// actor + masked softmax(T=3). 4 MFLOP total -> one block, one row/thread.
// ---------------------------------------------------------------------------
__global__ void head_kernel(const bf16_t* __restrict__ hroot,
                            const float* __restrict__ W1, const float* __restrict__ b1,
                            const float* __restrict__ W2, const float* __restrict__ b2,
                            const float* __restrict__ W_ih, const float* __restrict__ b_lstm,
                            const float* __restrict__ actor_W, const float* __restrict__ actor_b,
                            const float* __restrict__ vm, float* __restrict__ out) {
    int b = threadIdx.x;                   // 0..127
    float h[64], t1[64], f[64], hg[64];
    for (int k = 0; k < 64; ++k) h[k] = (float)hroot[b * 64 + k];
    for (int j = 0; j < 64; ++j) {
        float s = b1[j];
        for (int k = 0; k < 64; ++k) s += h[k] * W1[k * 64 + j];
        t1[j] = fmaxf(s, 0.0f);
    }
    for (int j = 0; j < 64; ++j) {
        float s = b2[j];
        for (int k = 0; k < 64; ++k) s += t1[k] * W2[k * 64 + j];
        f[j] = s;
    }
    for (int j = 0; j < 64; ++j) {
        float gi = b_lstm[j], gg = b_lstm[128 + j], go = b_lstm[192 + j];
        for (int k = 0; k < 64; ++k) {
            float fk = f[k];
            gi += fk * W_ih[k * 256 + j];
            gg += fk * W_ih[k * 256 + 128 + j];
            go += fk * W_ih[k * 256 + 192 + j];
        }
        float cg = sigf(gi) * tanhf(gg);
        hg[j] = sigf(go) * tanhf(cg);
    }
    float logits[20], mx = -1e30f;
    for (int o = 0; o < 20; ++o) {
        float s = 0.0f;
        for (int k = 0; k < 64; ++k) s += f[k]  * actor_W[k * 20 + o];
        for (int k = 0; k < 64; ++k) s += hg[k] * actor_W[(64 + k) * 20 + o];
        float l = (__logf(vm[o]) + s * vm[o] + actor_b[o] * vm[o]) * (1.0f / 3.0f);
        logits[o] = l;
        mx = fmaxf(mx, l);
    }
    float denom = 0.0f;
    for (int o = 0; o < 20; ++o) { float e = __expf(logits[o] - mx); logits[o] = e; denom += e; }
    for (int o = 0; o < 20; ++o) out[b * 20 + o] = logits[o] / denom;
}

// ---------------------------------------------------------------------------
extern "C" void kernel_launch(void* const* d_in, const int* in_sizes, int n_in,
                              void* d_out, int out_size, void* d_ws, size_t ws_size,
                              hipStream_t stream) {
    const int*   x_idx   = (const int*)  d_in[0];
    const float* vm      = (const float*)d_in[1];
    const float* Wx      = (const float*)d_in[2];
    const float* Ua      = (const float*)d_in[3];
    const float* Ub      = (const float*)d_in[4];
    const float* b_tree  = (const float*)d_in[5];
    const float* W_ih    = (const float*)d_in[6];
    /* d_in[7] = W_hh, unused: h0 == 0 */
    const float* b_lstm  = (const float*)d_in[8];
    const float* W1      = (const float*)d_in[9];
    const float* b1      = (const float*)d_in[10];
    const float* W2      = (const float*)d_in[11];
    const float* b2      = (const float*)d_in[12];
    const float* actor_W = (const float*)d_in[13];
    const float* actor_b = (const float*)d_in[14];
    float* out = (float*)d_out;

    // workspace layout (parity ping-pong; parity1 max M = 262144, parity0 = 131072)
    char* ws = (char*)d_ws;
    const size_t M0 = (size_t)131072 * HDIM;   // elems, even-depth levels
    const size_t M1 = (size_t)262144 * HDIM;   // elems, odd-depth levels
    bf16_t* packW = (bf16_t*)ws;                              // 80 KB
    size_t off = 256 * 1024;
    bf16_t* hbuf[2];
    float*  cbuf[2];
    hbuf[0] = (bf16_t*)(ws + off); off += M0 * sizeof(bf16_t);
    hbuf[1] = (bf16_t*)(ws + off); off += M1 * sizeof(bf16_t);
    cbuf[0] = (float*) (ws + off); off += M0 * sizeof(float);
    cbuf[1] = (float*) (ws + off); off += M1 * sizeof(float);

    pack_weights<<<160, 256, 0, stream>>>(Ua, Ub, packW);
    tree_leaf<<<65536, 256, 0, stream>>>(x_idx, Wx, b_tree, hbuf[1], cbuf[1]);
    for (int d = 10; d >= 0; --d) {
        int n = 1 << d;
        int blocks = (BATCH * n) / 32;
        tree_level_wmma<<<blocks, 128, 0, stream>>>(d, x_idx, Wx, b_tree, packW,
                                                    hbuf[(d + 1) & 1], cbuf[(d + 1) & 1],
                                                    hbuf[d & 1], cbuf[d & 1]);
    }
    head_kernel<<<1, 128, 0, stream>>>(hbuf[0], W1, b1, W2, b2,
                                       W_ih, b_lstm, actor_W, actor_b, vm, out);
}
